// CrossSpatialAttention_13889924235915
// MI455X (gfx1250) — matrix operand order repaired and checked by reading, not compiled, not run on hardware
//
#include <hip/hip_runtime.h>
#include <hip/hip_bf16.h>

#define NSP   8192
#define CIN   256
#define MIDC  64
#define BATCH 2
#define KV_LD 72   // padded row stride (elements) for LDS-staged K/V tiles

typedef __attribute__((ext_vector_type(16))) __bf16 v16bf;
typedef __attribute__((ext_vector_type(8)))  __bf16 v8bf;
typedef __attribute__((ext_vector_type(8)))  float  v8f;
typedef int v4i __attribute__((vector_size(16)));                 // matches builtin param
typedef __attribute__((address_space(1))) v4i* gv4i_p;            // global int4*
typedef __attribute__((address_space(3))) v4i* lv4i_p;            // LDS int4*

#ifndef __has_builtin
#define __has_builtin(x) 0
#endif
#if __has_builtin(__builtin_amdgcn_global_load_async_to_lds_b128)
#define HAVE_ASYNC_LDS 1
#else
#define HAVE_ASYNC_LDS 0
#endif

__device__ __forceinline__ v8f wmma_bf16(v16bf a, v16bf b, v8f c) {
  // emits v_wmma_f32_16x16x32_bf16
  return __builtin_amdgcn_wmma_f32_16x16x32_bf16(false, a, false, b, (short)0, c, false, false);
}

// A-fragment (16x32, 16-bit): lane<16 -> M=lane, K = k0..k0+15;
// lane>=16 -> M=lane-16, K = k0+16..k0+31
__device__ __forceinline__ v16bf load_afrag(const __bf16* rowptr, int k0, int lane) {
  int kb = k0 + ((lane & 16) ? 16 : 0);
  v8bf lo = *(const v8bf*)(rowptr + kb);
  v8bf hi = *(const v8bf*)(rowptr + kb + 8);
  v16bf r;
#pragma unroll
  for (int i = 0; i < 8; ++i) { r[i] = lo[i]; r[i + 8] = hi[i]; }
  return r;
}

// B-fragment (32x16, 16-bit): per-lane col N = n0+(lane&15); 16 consecutive K
// starting at k0 (lanes 0-15) or k0+16 (lanes 16-31). buf[n*ld + k] layout.
__device__ __forceinline__ v16bf load_bfrag(const __bf16* buf, size_t ld, int n0, int k0, int lane) {
  int n = n0 + (lane & 15);
  int ks = k0 + ((lane & 16) ? 16 : 0);
  return *(const v16bf*)(buf + (size_t)n * ld + ks);
}

// 16-byte async global->LDS copy (falls back to sync copy through VGPRs)
__device__ __forceinline__ void async_copy16(const __bf16* src, __bf16* lds_dst) {
#if HAVE_ASYNC_LDS
  __builtin_amdgcn_global_load_async_to_lds_b128((gv4i_p)src, (lv4i_p)lds_dst, 0, 0);
#else
  *(v8bf*)lds_dst = *(const v8bf*)src;
#endif
}

__device__ __forceinline__ void wait_async0() {
#if HAVE_ASYNC_LDS
#if __has_builtin(__builtin_amdgcn_s_wait_asynccnt)
  __builtin_amdgcn_s_wait_asynccnt(0);
#else
  asm volatile("s_wait_asynccnt 0" ::: "memory");
#endif
#endif
}

// ---------------------------------------------------------------------------
// Kernel 0: fold BN into 1x1 conv weights, convert to bf16.
// Aw = [Af | Ag | Ah], each [64][256]; cb = [cf | cg | ch], each [64]; WvB [256][64]
// ---------------------------------------------------------------------------
__global__ void prep_kernel(const float* __restrict__ Wf, const float* __restrict__ bf,
                            const float* __restrict__ gamf, const float* __restrict__ betf,
                            const float* __restrict__ mf, const float* __restrict__ vf,
                            const float* __restrict__ Wg, const float* __restrict__ bg,
                            const float* __restrict__ gamg, const float* __restrict__ betg,
                            const float* __restrict__ mg, const float* __restrict__ vg,
                            const float* __restrict__ Wh, const float* __restrict__ bh,
                            const float* __restrict__ Wv,
                            __bf16* __restrict__ Aw, float* __restrict__ cb,
                            __bf16* __restrict__ WvB) {
  int i = blockIdx.x * blockDim.x + threadIdx.x;
  if (i < MIDC * CIN) {
    int c = i / CIN;
    float invf = gamf[c] * rsqrtf(vf[c] + 1e-5f);
    float invg = gamg[c] * rsqrtf(vg[c] + 1e-5f);
    Aw[i]                  = (__bf16)(Wf[i] * invf);
    Aw[MIDC * CIN + i]     = (__bf16)(Wg[i] * invg);
    Aw[2 * MIDC * CIN + i] = (__bf16)(Wh[i]);
    if ((i % CIN) == 0) {
      cb[c]       = bf[c] * invf + betf[c] - mf[c] * invf;
      cb[64 + c]  = bg[c] * invg + betg[c] - mg[c] * invg;
      cb[128 + c] = bh[c];
    }
    WvB[i] = (__bf16)Wv[i];  // [256][64] row-major, same element count
  }
}

// ---------------------------------------------------------------------------
// Kernel 0b: x [B][256][8192] f32 -> xT [B][8192][256] bf16 (LDS tile transpose)
// ---------------------------------------------------------------------------
__global__ __launch_bounds__(256) void transpose_x(const float* __restrict__ x,
                                                   __bf16* __restrict__ xT) {
  __shared__ __bf16 tile[32][33];
  int n0 = blockIdx.x * 32, k0 = blockIdx.y * 32, b = blockIdx.z;
  const float* xb = x + (size_t)b * CIN * NSP;
  __bf16* xTb = xT + (size_t)b * NSP * CIN;
#pragma unroll
  for (int j = 0; j < 4; ++j) {
    int k = k0 + threadIdx.y + j * 8;
    tile[threadIdx.y + j * 8][threadIdx.x] = (__bf16)xb[(size_t)k * NSP + n0 + threadIdx.x];
  }
  __syncthreads();
#pragma unroll
  for (int j = 0; j < 4; ++j) {
    int n = n0 + threadIdx.y + j * 8;
    xTb[(size_t)n * CIN + k0 + threadIdx.x] = tile[threadIdx.x][threadIdx.y + j * 8];
  }
}

// ---------------------------------------------------------------------------
// Kernel 1: projections f, g, h. One wave = 16 spatial rows, all 64 mid-channels.
// fT/gT stored [B][N][64] bf16 (f scaled by MID^-0.5, both ReLU'd), h [B][64][N].
// ---------------------------------------------------------------------------
__global__ __launch_bounds__(256) void proj_kernel(const __bf16* __restrict__ xT,
                                                   const __bf16* __restrict__ Aw,
                                                   const float* __restrict__ cb,
                                                   __bf16* __restrict__ fT,
                                                   __bf16* __restrict__ gT,
                                                   __bf16* __restrict__ hC) {
  int lane = threadIdx.x & 31, wave = threadIdx.x >> 5;
  int tile = blockIdx.x * 8 + wave;            // 0..1023
  int b = tile >> 9;
  int n0 = (tile & 511) * 16;
  int lo = lane & 15, hi8 = (lane & 16) ? 8 : 0;

  const __bf16* xrow = xT + ((size_t)b * NSP + n0 + lo) * CIN;
  v16bf xa[8];
#pragma unroll
  for (int kk = 0; kk < 8; ++kk) xa[kk] = load_afrag(xrow, kk * 32, lane);

  const __bf16* Af = Aw;
  const __bf16* Ag = Aw + MIDC * CIN;
  const __bf16* Ah = Aw + 2 * MIDC * CIN;
  __bf16* fTb = fT + (size_t)b * NSP * MIDC;
  __bf16* gTb = gT + (size_t)b * NSP * MIDC;
  __bf16* hCb = hC + (size_t)b * MIDC * NSP;

  for (int ct = 0; ct < 4; ++ct) {
    v8f af = {0,0,0,0,0,0,0,0}, ag = {0,0,0,0,0,0,0,0}, ah = {0,0,0,0,0,0,0,0};
#pragma unroll
    for (int kk = 0; kk < 8; ++kk) {
      af = wmma_bf16(xa[kk], load_bfrag(Af, CIN, ct * 16, kk * 32, lane), af);
      ag = wmma_bf16(xa[kk], load_bfrag(Ag, CIN, ct * 16, kk * 32, lane), ag);
      ah = wmma_bf16(xa[kk], load_bfrag(Ah, CIN, ct * 16, kk * 32, lane), ah);
    }
    int c = ct * 16 + lo;
    float bfv = cb[c], bgv = cb[64 + c], bhv = cb[128 + c];
#pragma unroll
    for (int r = 0; r < 8; ++r) {
      int n = n0 + r + hi8;
      fTb[(size_t)n * MIDC + c] = (__bf16)(fmaxf(af[r] + bfv, 0.0f) * 0.125f);
      gTb[(size_t)n * MIDC + c] = (__bf16)(fmaxf(ag[r] + bgv, 0.0f));
      hCb[(size_t)c * NSP + n]  = (__bf16)(ah[r] + bhv);
    }
  }
}

// ---------------------------------------------------------------------------
// Kernel 2: fused flash attention. One wave = 16 query rows; block = 128 rows.
// K/V 64-column blocks are double-buffered in LDS via async global->LDS copies
// shared by all 8 waves (8x L2 traffic reduction). zT out [B][N][64] bf16.
// ---------------------------------------------------------------------------
__device__ __forceinline__ void stage_kv(const __bf16* __restrict__ gTb,
                                         const __bf16* __restrict__ hb, int m0,
                                         __bf16* kg, __bf16* hs, int tid) {
#pragma unroll
  for (int it = 0; it < 2; ++it) {
    int i = tid + it * 256;            // 0..511 -> 512 x 16B chunks per array
    int row = i >> 3;                  // 0..63 (key row for kg, channel for hs)
    int part = (i & 7) * 8;            // element offset within row
    async_copy16(gTb + (size_t)(m0 + row) * MIDC + part, kg + row * KV_LD + part);
    async_copy16(hb + (size_t)row * NSP + m0 + part,     hs + row * KV_LD + part);
  }
}

__global__ __launch_bounds__(256) void attn_kernel(const __bf16* __restrict__ fT,
                                                   const __bf16* __restrict__ gT,
                                                   const __bf16* __restrict__ hC,
                                                   __bf16* __restrict__ zT) {
  __shared__ __align__(16) __bf16 Kg[2][64 * KV_LD];   // 2 x 9216 B
  __shared__ __align__(16) __bf16 Hs[2][64 * KV_LD];   // 2 x 9216 B
  __shared__ __align__(16) __bf16 ldsP[8 * 16 * 72];   // per-wave 16x64 P tile

  int tid  = threadIdx.x;
  int lane = tid & 31, wave = tid >> 5;
  int strip = blockIdx.x * 8 + wave;          // 0..1023; whole block same batch
  int b = strip >> 9;
  int n0 = (strip & 511) * 16;
  int lo = lane & 15, hi8 = (lane & 16) ? 8 : 0;

  const __bf16* fTb = fT + (size_t)b * NSP * MIDC;
  const __bf16* gTb = gT + (size_t)b * NSP * MIDC;
  const __bf16* hb  = hC + (size_t)b * MIDC * NSP;
  __bf16* zTb = zT + (size_t)b * NSP * MIDC;
  __bf16* myP = &ldsP[wave * 16 * 72];

  // Q fragments: rows n0..n0+15, K=0..63 (scale already folded into f)
  const __bf16* qrow = fTb + (size_t)(n0 + lo) * MIDC;
  v16bf qa0 = load_afrag(qrow, 0, lane);
  v16bf qa1 = load_afrag(qrow, 32, lane);

  v8f acc[4];
#pragma unroll
  for (int t = 0; t < 4; ++t) acc[t] = (v8f){0,0,0,0,0,0,0,0};
  float mrow[8], lrow[8];
#pragma unroll
  for (int r = 0; r < 8; ++r) { mrow[r] = -3.0e38f; lrow[r] = 0.0f; }

  stage_kv(gTb, hb, 0, Kg[0], Hs[0], tid);

  for (int it = 0; it < NSP / 64; ++it) {
    int m0 = it * 64;
    int pb = it & 1;
    wait_async0();          // this wave's staged copies have landed in LDS
    __syncthreads();        // all waves' copies landed; prior compute finished
    if (it + 1 < NSP / 64)  // prefetch next block into the other buffer
      stage_kv(gTb, hb, m0 + 64, Kg[pb ^ 1], Hs[pb ^ 1], tid);
    const __bf16* kg = Kg[pb];
    const __bf16* hs = Hs[pb];

    // ---- S = Q K^T  (16 x 64, fp32) ----
    v8f s[4];
#pragma unroll
    for (int t = 0; t < 4; ++t) {
      v8f st = {0,0,0,0,0,0,0,0};
      st = wmma_bf16(qa0, load_bfrag(kg, KV_LD, t * 16, 0,  lane), st);
      st = wmma_bf16(qa1, load_bfrag(kg, KV_LD, t * 16, 32, lane), st);
      s[t] = st;
    }
    // ---- online softmax (per-lane slot r = row r+hi8; shfl within 16-lane half) ----
#pragma unroll
    for (int r = 0; r < 8; ++r) {
      float v = fmaxf(fmaxf(s[0][r], s[1][r]), fmaxf(s[2][r], s[3][r]));
      v = fmaxf(v, __shfl_xor(v, 1, 32));
      v = fmaxf(v, __shfl_xor(v, 2, 32));
      v = fmaxf(v, __shfl_xor(v, 4, 32));
      v = fmaxf(v, __shfl_xor(v, 8, 32));
      float mn = fmaxf(mrow[r], v);
      float es = __expf(mrow[r] - mn);
      mrow[r] = mn;
      float p0 = __expf(s[0][r] - mn); s[0][r] = p0;
      float p1 = __expf(s[1][r] - mn); s[1][r] = p1;
      float p2 = __expf(s[2][r] - mn); s[2][r] = p2;
      float p3 = __expf(s[3][r] - mn); s[3][r] = p3;
      float rs = (p0 + p1) + (p2 + p3);
      rs += __shfl_xor(rs, 1, 32);
      rs += __shfl_xor(rs, 2, 32);
      rs += __shfl_xor(rs, 4, 32);
      rs += __shfl_xor(rs, 8, 32);
      lrow[r] = lrow[r] * es + rs;
      acc[0][r] *= es; acc[1][r] *= es; acc[2][r] *= es; acc[3][r] *= es;
    }
    // ---- stage P (D-layout) to LDS, reload as A-fragments ----
#pragma unroll
    for (int t = 0; t < 4; ++t)
#pragma unroll
      for (int r = 0; r < 8; ++r)
        myP[(r + hi8) * 72 + t * 16 + lo] = (__bf16)s[t][r];
    asm volatile("s_wait_dscnt 0" ::: "memory");   // wave-local LDS RAW
    // ---- acc += P * h^T  (K = 64 local columns of this block) ----
#pragma unroll
    for (int mc = 0; mc < 2; ++mc) {
      v16bf pa = load_afrag(myP + lo * 72, mc * 32, lane);
#pragma unroll
      for (int ct = 0; ct < 4; ++ct)
        acc[ct] = wmma_bf16(pa, load_bfrag(hs, KV_LD, ct * 16, mc * 32, lane), acc[ct]);
    }
  }

  // ---- normalize and store zT [N][64] bf16 ----
  float rl[8];
#pragma unroll
  for (int r = 0; r < 8; ++r) rl[r] = 1.0f / lrow[r];
#pragma unroll
  for (int t = 0; t < 4; ++t)
#pragma unroll
    for (int r = 0; r < 8; ++r)
      zTb[(size_t)(n0 + r + hi8) * MIDC + t * 16 + lo] = (__bf16)(acc[t][r] * rl[r]);
}

// ---------------------------------------------------------------------------
// Kernel 3: out = Wv * z + bv + x  (M=256 out-ch, N=8192 spatial, K=64), fp32 out
// ---------------------------------------------------------------------------
__global__ __launch_bounds__(256) void outconv_kernel(const __bf16* __restrict__ WvB,
                                                      const float* __restrict__ bv,
                                                      const __bf16* __restrict__ zT,
                                                      const float* __restrict__ x,
                                                      float* __restrict__ out) {
  int lane = threadIdx.x & 31, wave = threadIdx.x >> 5;
  int tile = blockIdx.x * 8 + wave;           // 0..16383
  int b = tile >> 13;
  int t = tile & 8191;
  int o0 = (t >> 9) * 16;
  int n0 = (t & 511) * 16;
  int lo = lane & 15, hi8 = (lane & 16) ? 8 : 0;

  const __bf16* wrow = WvB + (size_t)(o0 + lo) * MIDC;
  const __bf16* zTb = zT + (size_t)b * NSP * MIDC;
  v8f acc = {0,0,0,0,0,0,0,0};
#pragma unroll
  for (int kk = 0; kk < 2; ++kk)
    acc = wmma_bf16(load_afrag(wrow, kk * 32, lane),
                    load_bfrag(zTb, MIDC, n0, kk * 32, lane), acc);
#pragma unroll
  for (int r = 0; r < 8; ++r) {
    int o = o0 + r + hi8;
    size_t idx = ((size_t)b * CIN + o) * NSP + n0 + lo;
    out[idx] = acc[r] + bv[o] + x[idx];
  }
}

// ---------------------------------------------------------------------------
extern "C" void kernel_launch(void* const* d_in, const int* in_sizes, int n_in,
                              void* d_out, int out_size, void* d_ws, size_t ws_size,
                              hipStream_t stream) {
  const float* x    = (const float*)d_in[0];
  const float* Wf   = (const float*)d_in[1];
  const float* bf   = (const float*)d_in[2];
  const float* gamf = (const float*)d_in[3];
  const float* betf = (const float*)d_in[4];
  const float* mf   = (const float*)d_in[5];
  const float* vf   = (const float*)d_in[6];
  const float* Wg   = (const float*)d_in[7];
  const float* bg   = (const float*)d_in[8];
  const float* gamg = (const float*)d_in[9];
  const float* betg = (const float*)d_in[10];
  const float* mg   = (const float*)d_in[11];
  const float* vg   = (const float*)d_in[12];
  const float* Wh   = (const float*)d_in[13];
  const float* bh   = (const float*)d_in[14];
  const float* Wv   = (const float*)d_in[15];
  const float* bv   = (const float*)d_in[16];
  float* out = (float*)d_out;

  char* ws = (char*)d_ws;
  __bf16* Aw  = (__bf16*)(ws);                       //  98304 B  [3][64][256] bf16
  float*  cb  = (float*)(ws + 98304);                //    768 B  [3][64] f32
  __bf16* WvB = (__bf16*)(ws + 99328);               //  32768 B  [256][64] bf16
  __bf16* xT  = (__bf16*)(ws + 132096);              // 8388608 B [2][8192][256] bf16
  __bf16* fT  = (__bf16*)(ws + 8520704);             // 2097152 B [2][8192][64]
  __bf16* gT  = (__bf16*)(ws + 10617856);            // 2097152 B
  __bf16* hC  = (__bf16*)(ws + 12715008);            // 2097152 B [2][64][8192]
  __bf16* zT  = (__bf16*)(ws + 14812160);            // 2097152 B [2][8192][64]

  prep_kernel<<<64, 256, 0, stream>>>(Wf, bf, gamf, betf, mf, vf,
                                      Wg, bg, gamg, betg, mg, vg,
                                      Wh, bh, Wv, Aw, cb, WvB);
  transpose_x<<<dim3(NSP / 32, CIN / 32, BATCH), dim3(32, 8), 0, stream>>>(x, xT);
  proj_kernel<<<128, 256, 0, stream>>>(xT, Aw, cb, fT, gT, hC);
  attn_kernel<<<128, 256, 0, stream>>>(fT, gT, hC, zT);
  outconv_kernel<<<2048, 256, 0, stream>>>(WvB, bv, zT, x, out);
}
